// Sac4DplWithNnModule_7352984011052
// MI455X (gfx1250) — matrix operand order repaired and checked by reading, not compiled, not run on hardware
//
#include <hip/hip_runtime.h>
#include <hip/hip_bf16.h>

// ---------------------------------------------------------------------------
// XAJ hydrological scan with embedded evap-MLP (8 -> 16 -> 8 -> 1) on WMMA.
// T=730 sequential steps, B=16384 independent basins. One wave32 owns 32
// basins and carries soil-moisture state in registers across all timesteps.
//
// MLP mapping (wave32, v_wmma_f32_16x16x32_f16, two 16-basin tiles/wave):
//   L1: H1 = feats(16x8 pad K->32) @ W1(8x16) + b1   (bias in WMMA C operand)
//   L2: transposed: D2 = W2^T(8x16 pad) @ H1relu^T(16x16) + b2 (bias in C)
//       -> lane n (basin) holds H2[0..7][basin] in its 8 acc VGPRs
//   L3: per-lane dot with W3 (no cross-lane traffic)
//
// Zero-padding discipline: bW1 halves 8..15 == 0 and aW2 lanes 16..31 == 0,
// so the "extra" halves/lanes of the A and B2 operands may hold junk
// (duplicates) -- every per-step operand build is branch-free, all statics
// (incl. their cross-lane shuffles) are hoisted out of the time loop.
// ---------------------------------------------------------------------------

typedef __attribute__((ext_vector_type(16))) _Float16 v16h;
typedef __attribute__((ext_vector_type(8)))  float    v8f;
typedef __attribute__((ext_vector_type(2)))  _Float16 h2_t;

#define T_STEPS 730
#define NBASIN  16384
#define EPS     1e-5f

__device__ __forceinline__ float sigmoidf_fast(float x) {
    return 1.0f / (1.0f + __expf(-x));
}

__global__ __launch_bounds__(32)
void xaj_dpl_scan_kernel(const float* __restrict__ p_and_e,  // [T,B,2]
                         const float* __restrict__ kc_p,     // [B]
                         const float* __restrict__ um_p,
                         const float* __restrict__ lm_p,
                         const float* __restrict__ dm_p,
                         const float* __restrict__ bexp_p,
                         const float* __restrict__ im_p,
                         const float* __restrict__ c_p,
                         const float* __restrict__ w0_p,
                         const float* __restrict__ W1_p,     // [8,16]
                         const float* __restrict__ b1_p,     // [16]
                         const float* __restrict__ W2_p,     // [16,8]
                         const float* __restrict__ b2_p,     // [8]
                         const float* __restrict__ W3_p,     // [8]
                         const float* __restrict__ b3_p,     // [1]
                         float* __restrict__ out)            // [T,B,4]
{
    const int lane = threadIdx.x;               // 0..31
    const int j    = blockIdx.x * 32 + lane;    // this lane's basin

    // ---- static per-basin parameters --------------------------------------
    const float kc   = kc_p[j];
    const float um   = um_p[j];
    const float lm   = lm_p[j];
    const float dm   = dm_p[j];
    const float bexp = bexp_p[j];
    const float im   = im_p[j];
    const float cc   = c_p[j];
    float       w    = w0_p[j];                 // recurrent state

    const float wm      = um + lm + dm;
    const float pb1     = 1.0f + bexp;
    const float wmm     = wm * pb1;
    const float inv_b1  = 1.0f / pb1;
    const float inv_wm  = 1.0f / wm;
    const float inv_wmm = 1.0f / wmm;
    const float wm_eps  = wm - EPS;

    const _Float16 kc_h = (_Float16)kc;
    const _Float16 um_h = (_Float16)um;
    const _Float16 lm_h = (_Float16)lm;
    const _Float16 dm_h = (_Float16)dm;
    const _Float16 c_h  = (_Float16)cc;

    // ---- weights in WMMA layouts (branch-free, clamped addressing) --------
    // B1 (W1, 32x16, K=8 used): lane%16 = col n, halves 0..7 = K 0..7.
    // Halves 8..15 MUST be zero (annihilates junk K=8..15 halves of A).
    v16h bW1 = {};
    #pragma unroll
    for (int h = 0; h < 8; ++h)
        bW1[h] = (lane < 16) ? (_Float16)W1_p[h * 16 + (lane & 15)] : (_Float16)0.0f;

    // A2 (W2^T padded 16x32, K=16 used): row m = lane%16, lanes 0..15 halves
    // 0..15 = K = 0..15 (contiguous order). Rows 8..15 are harmless duplicates
    // (their D2 rows land in lanes 16..31, never consumed). Lanes 16..31
    // MUST be zero (annihilates junk K=16..31 lanes of B2).
    v16h aW2 = {};
    {
        #pragma unroll
        for (int h = 0; h < 16; ++h)
            aW2[h] = (lane < 16) ? (_Float16)W2_p[h * 8 + (lane & 7)] : (_Float16)0.0f;
    }

    // ---- biases folded into WMMA C accumulators ---------------------------
    // C1[m][n] = b1[n]: D-layout => every acc vgpr of a lane = b1[lane%16].
    const float b1v = b1_p[lane & 15];
    v8f cbias1;
    #pragma unroll
    for (int v = 0; v < 8; ++v) cbias1[v] = b1v;
    // C2[h2][basin] = b2[h2]: transposed D2 layout => acc vgpr v = b2[v].
    v8f cbias2;
    float w3r[8];
    #pragma unroll
    for (int v = 0; v < 8; ++v) { cbias2[v] = b2_p[v]; w3r[v] = W3_p[v]; }
    const float b3v = b3_p[0];

    // ---- A operands: static halves hoisted out of the time loop -----------
    // A1 layout (16x32 f16): lane%16 = row, halves 0..7 = K 0..7 in feature
    // order {kc,um,lm,dm,c,w0c,prcp,pet}. Lanes 16..31 hold duplicates
    // (multiplied by bW1's zero halves). Halves 8..15 stay zero forever.
    const int src = (lane & 15) + 16;           // tile-1 source lane
    v16h a0 = {};
    a0[0] = kc_h; a0[1] = um_h; a0[2] = lm_h; a0[3] = dm_h; a0[4] = c_h;

    v16h a1 = {};
    {
        h2_t p01 = {kc_h, um_h};
        h2_t p23 = {lm_h, dm_h};
        h2_t p4x = {c_h, (_Float16)0.0f};
        int s01 = __shfl(__builtin_bit_cast(int, p01), src, 32);
        int s23 = __shfl(__builtin_bit_cast(int, p23), src, 32);
        int s4x = __shfl(__builtin_bit_cast(int, p4x), src, 32);
        h2_t r01 = __builtin_bit_cast(h2_t, s01);
        h2_t r23 = __builtin_bit_cast(h2_t, s23);
        h2_t r4x = __builtin_bit_cast(h2_t, s4x);
        a1[0] = r01[0]; a1[1] = r01[1];
        a1[2] = r23[0]; a1[3] = r23[1];
        a1[4] = r4x[0];
    }

    // H1relu staging for the transpose into B2 layout: [tile][basin_row*16+h1]
    __shared__ alignas(32) _Float16 lds_h1[2][256];

    const float2* __restrict__ pe2  = (const float2*)p_and_e;
    float4* __restrict__       out4 = (float4*)out;

    for (int t = 0; t < T_STEPS; ++t) {
        const float2 pp = pe2[(size_t)t * NBASIN + j];
        {   // next-step prefetch (clamped address) -> global_prefetch_b8
            const int tn = (t + 1 < T_STEPS) ? (t + 1) : t;
            __builtin_prefetch(&pe2[(size_t)tn * NBASIN + j], 0, 1);
        }

        const float prcp = fmaxf(pp.x, 0.0f);
        const float pet  = fmaxf(pp.y, 0.0f);
        const float w0c  = fminf(fmaxf(w, 0.0f), wm_eps);

        // ---- dynamic features: 3 halves per tile, 2 shuffles, no branches -
        const _Float16 w0c_h = (_Float16)w0c;
        const _Float16 pr_h  = (_Float16)prcp;
        const _Float16 pt_h  = (_Float16)pet;

        a0[5] = w0c_h; a0[6] = pr_h; a0[7] = pt_h;

        h2_t qd0 = {w0c_h, pr_h};
        h2_t qd1 = {pt_h, (_Float16)0.0f};
        int t0 = __shfl(__builtin_bit_cast(int, qd0), src, 32);
        int t1 = __shfl(__builtin_bit_cast(int, qd1), src, 32);
        h2_t r0 = __builtin_bit_cast(h2_t, t0);
        h2_t r1 = __builtin_bit_cast(h2_t, t1);
        a1[5] = r0[0]; a1[6] = r0[1]; a1[7] = r1[0];

        // ---- layer 1: H1pre = feats @ W1 + b1 (bias via C operand) --------
        v8f d1t0 = __builtin_amdgcn_wmma_f32_16x16x32_f16(
            false, a0, false, bW1, (short)0, cbias1, false, false);
        v8f d1t1 = __builtin_amdgcn_wmma_f32_16x16x32_f16(
            false, a1, false, bW1, (short)0, cbias1, false, false);

        // relu + f16, stage transposed operand in LDS:
        // D1 layout: lane = n + 16*(m>=8), vgpr v -> m = v + 8*(lane>=16)
        {
            const int nn    = lane & 15;
            const int mbase = (lane >> 4) * 8;
            #pragma unroll
            for (int v = 0; v < 8; ++v) {
                const int m = mbase + v;
                lds_h1[0][m * 16 + nn] = (_Float16)fmaxf(d1t0[v], 0.0f);
                lds_h1[1][m * 16 + nn] = (_Float16)fmaxf(d1t1[v], 0.0f);
            }
        }
        __syncthreads();   // single-wave WG: degenerates to wait/NOP

        // B2 layout (32x16, K=16 used): col = lane%16 (basin), halves 0..15 =
        // K = h1 index -> contiguous LDS row. Upper lanes read duplicates
        // (K=16..31), annihilated by aW2's zero lanes 16..31.
        v16h b2t0 = *(const v16h*)&lds_h1[0][(lane & 15) * 16];
        v16h b2t1 = *(const v16h*)&lds_h1[1][(lane & 15) * 16];
        __syncthreads();

        // ---- layer 2 (transposed): D2 = W2^T @ H1relu^T + b2 (via C) ------
        v8f d2t0 = __builtin_amdgcn_wmma_f32_16x16x32_f16(
            false, aW2, false, b2t0, (short)0, cbias2, false, false);
        v8f d2t1 = __builtin_amdgcn_wmma_f32_16x16x32_f16(
            false, aW2, false, b2t1, (short)0, cbias2, false, false);

        // ---- layer 3: per-lane dot (lane<16 holds one basin's 8 h2 vals) --
        float z0 = b3v, z1 = b3v;
        #pragma unroll
        for (int v = 0; v < 8; ++v) {
            z0 += w3r[v] * fmaxf(d2t0[v], 0.0f);
            z1 += w3r[v] * fmaxf(d2t1[v], 0.0f);
        }
        // route z back to owning lane: basins 16..31 live in z1 of lanes 0..15
        const float z1s = __shfl(z1, lane & 15, 32);
        const float z   = (lane < 16) ? z0 : z1s;

        // ---- XAJ hydrology (all 32 lanes active) --------------------------
        const float e      = sigmoidf_fast(z) * kc * pet;
        const float pd     = prcp - e;
        const float pe_net = fmaxf(pd, 0.0f);
        const float frac   = fminf(fmaxf(1.0f - w0c * inv_wm, EPS), 1.0f);
        const float a_t    = wmm * (1.0f - __powf(frac, inv_b1));
        const float r_full = pe_net - (wm - w0c);
        const float resid  = fminf(fmaxf(1.0f - (pe_net + a_t) * inv_wmm, 0.0f), 1.0f);
        float r = (pe_net + a_t < wmm) ? (r_full + wm * __powf(resid, pb1))
                                       : r_full;
        r = fmaxf(r, 0.0f);
        const float rim = pe_net * im;
        w = fminf(fmaxf(w0c + pd - r, 0.0f), wm_eps);

        float4 o;
        o.x = r; o.y = rim; o.z = e; o.w = pe_net;
        out4[(size_t)t * NBASIN + j] = o;
    }
}

extern "C" void kernel_launch(void* const* d_in, const int* in_sizes, int n_in,
                              void* d_out, int out_size, void* d_ws, size_t ws_size,
                              hipStream_t stream) {
    (void)in_sizes; (void)n_in; (void)out_size; (void)d_ws; (void)ws_size;
    const float* p_and_e = (const float*)d_in[0];
    const float* kc   = (const float*)d_in[1];
    const float* um   = (const float*)d_in[2];
    const float* lm   = (const float*)d_in[3];
    const float* dm   = (const float*)d_in[4];
    const float* bexp = (const float*)d_in[5];
    const float* im   = (const float*)d_in[6];
    const float* c    = (const float*)d_in[7];
    const float* w0   = (const float*)d_in[8];
    const float* W1   = (const float*)d_in[9];
    const float* b1   = (const float*)d_in[10];
    const float* W2   = (const float*)d_in[11];
    const float* b2   = (const float*)d_in[12];
    const float* W3   = (const float*)d_in[13];
    const float* b3   = (const float*)d_in[14];

    dim3 grid(NBASIN / 32);   // 512 single-wave workgroups -> spread over WGPs
    dim3 block(32);
    xaj_dpl_scan_kernel<<<grid, block, 0, stream>>>(
        p_and_e, kc, um, lm, dm, bexp, im, c, w0,
        W1, b1, W2, b2, W3, b3, (float*)d_out);
}
